// LSTM_37924561224180
// MI455X (gfx1250) — matrix operand order repaired and checked by reading, not compiled, not run on hardware
//
#include <hip/hip_runtime.h>

#define SEQ 512
#define BSZ 32
#define ISZ 1024
#define HSZ 1024
#define LNUM 2

typedef __attribute__((ext_vector_type(16))) __bf16 v16bf;
typedef __attribute__((ext_vector_type(8)))  float  v8f;
typedef __attribute__((ext_vector_type(4)))  unsigned v4u;

union FragU { v4u u[2]; v16bf v; };

__device__ __forceinline__ unsigned short f2bf(float f) {
  union { float f; unsigned u; } x; x.f = f;
  unsigned r = x.u + 0x7FFFu + ((x.u >> 16) & 1u);   // round-to-nearest-even
  return (unsigned short)(r >> 16);
}

__device__ __forceinline__ float sigmoidf(float x) {
  return 1.0f / (1.0f + __expf(-x));
}

// ---- prep: fp32 -> bf16 elementwise (for x) ----
__global__ void cvt_f32_bf16(const float* __restrict__ in,
                             unsigned short* __restrict__ out, int n) {
  int i = blockIdx.x * blockDim.x + threadIdx.x;
  int stride = gridDim.x * blockDim.x;
  for (; i < n; i += stride) out[i] = f2bf(in[i]);
}

// ---- prep: W[l][k][n] fp32 -> Wt[l][n][k] bf16 (LDS tile transpose) ----
__global__ __launch_bounds__(256) void transpose_w(const float* __restrict__ W,
                                                   unsigned short* __restrict__ Wt) {
  __shared__ unsigned short t[32][33];
  const int l  = blockIdx.z;
  const int n0 = blockIdx.x * 32;
  const int k0 = blockIdx.y * 32;
  const int tx = threadIdx.x & 31;
  const int ty = threadIdx.x >> 5;          // 0..7
  const float* Wl = W + (size_t)l * 2048 * 4096;
  unsigned short* Wtl = Wt + (size_t)l * 4096 * 2048;
  #pragma unroll
  for (int r = 0; r < 4; ++r) {
    int k = k0 + ty + r * 8;
    t[ty + r * 8][tx] = f2bf(Wl[(size_t)k * 4096 + n0 + tx]);
  }
  __syncthreads();
  #pragma unroll
  for (int r = 0; r < 4; ++r) {
    int n = n0 + ty + r * 8;
    Wtl[(size_t)n * 2048 + k0 + tx] = t[tx][ty + r * 8];
  }
}

// ---- prep: initial states ----
__global__ void init_state(const float* __restrict__ h0, const float* __restrict__ c0,
                           unsigned short* __restrict__ hping, float* __restrict__ c) {
  int i = blockIdx.x * blockDim.x + threadIdx.x;       // over L*B*H
  if (i >= LNUM * BSZ * HSZ) return;
  int l = i / (BSZ * HSZ);
  int rest = i - l * (BSZ * HSZ);
  hping[((size_t)l * 2 + 0) * (BSZ * HSZ) + rest] = f2bf(h0[i]);  // ping buffer 0
  c[i] = c0[i];
}

// ---- fused pipelined step ----
// Kernel k: WGs 0-63   -> layer0 at time t=k      (unit 0)
//           WGs 64-127 -> layer1 at time t=k-1    (unit 1)   [independent of unit 0]
// Each WG owns 16 hidden units; stages A = [x_t | h_prev] (32x2048 bf16, 128 KB)
// in LDS once, then 8 waves = (4 gates x 2 K-halves) stream W from L2 with zero
// fragment duplication. Split-K partials combined in LDS; gates fused in-WG.
#define AROW 2056  // ushorts/row: 4112 B stride -> 4-dword bank skew, conflict-free b128 frags

__global__ __launch_bounds__(256) void lstm_fused(
    const unsigned short* __restrict__ x0, const unsigned short* __restrict__ h0i,
    unsigned short* __restrict__ h0o, float* __restrict__ c0p,
    const unsigned short* __restrict__ W0, const float* __restrict__ b0,
    float* __restrict__ o0, float* __restrict__ hf0, float* __restrict__ cf0,
    const unsigned short* __restrict__ x1, const unsigned short* __restrict__ h1i,
    unsigned short* __restrict__ h1o, float* __restrict__ c1p,
    const unsigned short* __restrict__ W1, const float* __restrict__ b1,
    float* __restrict__ o1, float* __restrict__ hf1, float* __restrict__ cf1) {
  __shared__ unsigned short Alds[32 * AROW];         // ~131.6 KB
  __shared__ float z[4][32][16];                     // 8 KB

  const int unit = blockIdx.x >> 6;
  const unsigned short* xin  = unit ? x1 : x0;
  const unsigned short* hin  = unit ? h1i : h0i;
  unsigned short* hout       = unit ? h1o : h0o;
  float* c                   = unit ? c1p : c0p;
  const unsigned short* Wt   = unit ? W1 : W0;
  const float* bias          = unit ? b1 : b0;
  float* out_f32             = unit ? o1 : o0;
  float* hfin                = unit ? hf1 : hf0;
  float* cfin                = unit ? cf1 : cf0;
  if (xin == nullptr) return;                        // inactive pipeline slot

  const int tid  = threadIdx.x;
  const int lane = tid & 31;
  const int wave = tid >> 5;                         // 0..7
  const int r    = lane & 15;
  const int half = lane >> 4;
  const int gate = wave & 3;                         // 0..3
  const int kh   = wave >> 2;                        // K-half 0/1
  const int j0   = (blockIdx.x & 63) << 4;           // hidden-unit base
  const int n    = (gate << 10) + j0 + r;            // global gate column

  const float bv = bias[n];                          // hoisted; overlaps staging

  // --- cooperative A staging: rows = batch, cols 0-1023 = x, 1024-2047 = h_prev ---
  {
    const int col = tid << 3;                        // 8 bf16 = 16 B per thread per row
    const unsigned short* src = (col < 1024) ? (xin + col) : (hin + col - 1024);
    #pragma unroll 4
    for (int row = 0; row < 32; ++row) {
      *(v4u*)&Alds[row * AROW + col] = *(const v4u*)(src + row * 1024);
    }
  }
  __syncthreads();

  // --- K loop: each wave covers one gate n-tile over one 1024-wide K half ---
  v8f acc0 = {}, acc1 = {};
  const unsigned short* Bp = Wt + (size_t)n * 2048 + (kh << 10);
  // two running LDS pointers (rows r and 16+r); all unrolled deltas fit DS immediates
  const unsigned short* ap0 = &Alds[r * AROW + (kh << 10) + (half << 3)];
  const unsigned short* ap1 = ap0 + 16 * AROW;
  #pragma unroll 4
  for (int k0 = 0; k0 < 1024; k0 += 32) {
    FragU bfr, a0, a1;
    const unsigned short* Bb = Bp + k0 + half * 8;
    bfr.u[0] = *(const v4u*)(Bb);
    bfr.u[1] = *(const v4u*)(Bb + 16);
    a0.u[0] = *(const v4u*)(ap0);                    // ISA 16-bit A-frag K mapping
    a0.u[1] = *(const v4u*)(ap0 + 16);
    a1.u[0] = *(const v4u*)(ap1);
    a1.u[1] = *(const v4u*)(ap1 + 16);
    ap0 += 32;
    ap1 += 32;
    acc0 = __builtin_amdgcn_wmma_f32_16x16x32_bf16(
        false, a0.v, false, bfr.v, (short)0, acc0, false, false);
    acc1 = __builtin_amdgcn_wmma_f32_16x16x32_bf16(
        false, a1.v, false, bfr.v, (short)0, acc1, false, false);
  }

  // --- combine split-K partials (+bias) in LDS ---
  const int mb = half << 3;                          // D rows: m = mb + v (per 16-tile)
  if (kh == 0) {
    #pragma unroll
    for (int v = 0; v < 8; ++v) {
      z[gate][mb + v][r]      = acc0[v] + bv;
      z[gate][16 + mb + v][r] = acc1[v] + bv;
    }
  }
  __syncthreads();
  if (kh == 1) {
    #pragma unroll
    for (int v = 0; v < 8; ++v) {
      z[gate][mb + v][r]      += acc0[v];
      z[gate][16 + mb + v][r] += acc1[v];
    }
  }
  __syncthreads();

  // --- fused gates for the 16 units this WG owns ---
  #pragma unroll
  for (int idx = tid; idx < 512; idx += 256) {
    const int bb = idx >> 4;
    const int j  = idx & 15;
    const float fg = sigmoidf(z[0][bb][j]);
    const float gg = tanhf(z[1][bb][j]);
    const float ig = sigmoidf(sigmoidf(z[2][bb][j]));   // faithful double sigmoid
    const float og = sigmoidf(z[3][bb][j]);
    const int gi = bb * 1024 + j0 + j;
    const float cn = c[gi] * fg + gg * ig;
    const float hn = og * tanhf(cn);
    c[gi] = cn;
    hout[gi] = f2bf(hn);
    if (out_f32) out_f32[gi] = hn;
    if (hfin) hfin[gi] = hn;
    if (cfin) cfin[gi] = cn;
  }
}

extern "C" void kernel_launch(void* const* d_in, const int* in_sizes, int n_in,
                              void* d_out, int out_size, void* d_ws, size_t ws_size,
                              hipStream_t stream) {
  const float* x  = (const float*)d_in[0];   // [S][B][I]
  const float* h0 = (const float*)d_in[1];   // [L][B][H]
  const float* c0 = (const float*)d_in[2];   // [L][B][H]
  const float* W  = (const float*)d_in[3];   // [L][I+H][4H]
  const float* b  = (const float*)d_in[4];   // [L][4H]
  float* out = (float*)d_out;

  char* ws = (char*)d_ws;
  size_t off = 0;
  auto take = [&](size_t bytes) -> char* {
    char* p = ws + off;
    off = (off + bytes + 255) & ~(size_t)255;
    return p;
  };
  unsigned short* Wt    = (unsigned short*)take((size_t)LNUM * 4096 * 2048 * 2); // 32 MB
  unsigned short* xb    = (unsigned short*)take((size_t)SEQ * BSZ * ISZ * 2);    // 32 MB
  unsigned short* hping = (unsigned short*)take((size_t)LNUM * 2 * BSZ * HSZ * 2);
  float*          cbuf  = (float*)take((size_t)LNUM * BSZ * HSZ * 4);

  // --- prep ---
  cvt_f32_bf16<<<2048, 256, 0, stream>>>(x, xb, SEQ * BSZ * ISZ);
  transpose_w<<<dim3(4096 / 32, 2048 / 32, LNUM), 256, 0, stream>>>(W, Wt);
  {
    int nst = LNUM * BSZ * HSZ;
    init_state<<<(nst + 255) / 256, 256, 0, stream>>>(h0, c0, hping, cbuf);
  }

  const size_t BH = (size_t)BSZ * HSZ;
  float* outs = out;
  float* hfin = out + (size_t)SEQ * BH;
  float* cfin = hfin + (size_t)LNUM * BH;

  // --- software-pipelined scan: kernel k = layer0(k) + layer1(k-1) ---
  for (int k = 0; k <= SEQ; ++k) {
    const unsigned short *x0p = nullptr, *h0ip = nullptr;
    unsigned short *h0op = nullptr;
    float *c0pp = nullptr, *o0p = nullptr, *hf0p = nullptr, *cf0p = nullptr;
    const unsigned short *W0p = nullptr;
    const float *b0p = nullptr;

    const unsigned short *x1p = nullptr, *h1ip = nullptr;
    unsigned short *h1op = nullptr;
    float *c1pp = nullptr, *o1p = nullptr, *hf1p = nullptr, *cf1p = nullptr;
    const unsigned short *W1p = nullptr;
    const float *b1p = nullptr;

    if (k < SEQ) {                                   // unit 0: layer 0 at t=k
      const int t = k, pin = t & 1, pout = pin ^ 1;
      x0p  = xb + (size_t)t * BSZ * ISZ;
      h0ip = hping + (size_t)(0 * 2 + pin) * BH;
      h0op = hping + (size_t)(0 * 2 + pout) * BH;
      c0pp = cbuf + 0 * BH;
      W0p  = Wt + (size_t)0 * 4096 * 2048;
      b0p  = b + 0 * 4096;
      o0p  = nullptr;
      hf0p = (t == SEQ - 1) ? hfin + 0 * BH : nullptr;
      cf0p = (t == SEQ - 1) ? cfin + 0 * BH : nullptr;
    }
    if (k >= 1) {                                    // unit 1: layer 1 at t=k-1
      const int s = k - 1, pin = s & 1, pout = pin ^ 1;
      x1p  = hping + (size_t)(0 * 2 + pout) * BH;    // layer0's h from previous kernel
      h1ip = hping + (size_t)(1 * 2 + pin) * BH;
      h1op = hping + (size_t)(1 * 2 + pout) * BH;
      c1pp = cbuf + 1 * BH;
      W1p  = Wt + (size_t)1 * 4096 * 2048;
      b1p  = b + 1 * 4096;
      o1p  = outs + (size_t)s * BH;
      hf1p = (s == SEQ - 1) ? hfin + 1 * BH : nullptr;
      cf1p = (s == SEQ - 1) ? cfin + 1 * BH : nullptr;
    }

    lstm_fused<<<128, 256, 0, stream>>>(
        x0p, h0ip, h0op, c0pp, W0p, b0p, o0p, hf0p, cf0p,
        x1p, h1ip, h1op, c1pp, W1p, b1p, o1p, hf1p, cf1p);
  }
}
